// LSTMModel_63144609185880
// MI455X (gfx1250) — matrix operand order repaired and checked by reading, not compiled, not run on hardware
//
#include <hip/hip_runtime.h>
#include <hip/hip_bf16.h>

typedef __attribute__((ext_vector_type(16))) _Float16 v16h;
typedef __attribute__((ext_vector_type(8)))  float    v8f;
typedef __attribute__((ext_vector_type(4)))  float    v4f;

#define B_   2048
#define T_   512
#define H_   50
#define G_   200      // 4*H
#define BT   16       // batch tile = WMMA M
#define NW   8        // waves per block
#define NT   256      // threads per block (8 x wave32)
#define XS   516      // x_sh row stride (floats)  -> bank-spread
#define GS   20       // gates_t row stride (floats)
#define HS   68       // h16 row stride (halves)   -> bank-spread
#define H32S 52

__device__ __forceinline__ float sigmoidf_(float x) {
  return 1.0f / (1.0f + __expf(-x));
}
__device__ __forceinline__ float tanhf_(float x) {
  x = fminf(fmaxf(x, -10.0f), 10.0f);
  float e = __expf(2.0f * x);
  return (e - 1.0f) / (e + 1.0f);
}

__global__ __launch_bounds__(NT) void lstm_wmma_kernel(
    const float* __restrict__ x,   const float* __restrict__ Wih,
    const float* __restrict__ Whh, const float* __restrict__ bih,
    const float* __restrict__ bhh, const float* __restrict__ fcw,
    const float* __restrict__ fcb, float* __restrict__ out)
{
  __shared__ float    x_sh[BT * XS];      // all timesteps of this batch tile
  __shared__ float    gates_t[208 * GS];  // [n_gate][m_batch], padded
  __shared__ _Float16 h16[BT * HS];       // h in f16, K-padded to 64 (zeros)
  __shared__ float    h32[BT * H32S];     // final-step h in f32 for FC head
  __shared__ float    Wih_s[G_];
  __shared__ float    bias_s[G_];

  const int tid  = threadIdx.x;
  const int lane = tid & 31;
  const int w    = tid >> 5;      // wave id 0..7
  const int m    = lane & 15;     // M index inside tile
  const int hi   = lane >> 4;     // half-wave select
  const int b0   = blockIdx.x * BT;

  // ---- stage x for all 512 timesteps (only global traffic besides weights) ----
  for (int i = tid; i < BT * T_; i += NT) {
    int mm = i >> 9, tt = i & 511;
    x_sh[mm * XS + tt] = x[(b0 + mm) * T_ + tt];
  }
  for (int i = tid; i < G_; i += NT) {
    Wih_s[i]  = Wih[i];                 // W_ih is [200][1]
    bias_s[i] = bih[i] + bhh[i];
  }
  for (int i = tid; i < BT * HS; i += NT) h16[i] = (_Float16)0.0f;

  // ---- hoist W_hh^T B-fragments into registers (loop-invariant) ----
  // B layout (32x16, f16): lane%16 <-> N column, VGPR v packs an adjacent K pair (AMD table order).
  union BU { v16h h; _Float16 e[16]; };
  BU Bf[2][2];
  #pragma unroll
  for (int tt = 0; tt < 2; ++tt) {
    int tile = tt * NW + w;             // 0..15, tiles >=13 are dead (zero B)
    #pragma unroll
    for (int kf = 0; kf < 2; ++kf) {
      int g = tile * 16 + m;            // N (gate index)
      #pragma unroll
      for (int v = 0; v < 8; ++v) {
        #pragma unroll
        for (int p = 0; p < 2; ++p) {
          int j = kf * 32 + ((v < 4) ? 0 : 16) + (hi ? 8 : 0) + 2 * (v & 3) + p;
          float val = (tile < 13 && g < G_ && j < H_) ? Whh[g * H_ + j] : 0.0f;
          Bf[tt][kf].e[2 * v + p] = (_Float16)val;
        }
      }
    }
  }

  // ---- static per-thread (m,j) ownership so c stays in registers ----
  int emm[4], ejj[4];
  float c[4];
  #pragma unroll
  for (int r = 0; r < 4; ++r) {
    int idx = tid + r * NT;
    emm[r] = idx / H_;
    ejj[r] = idx % H_;
    c[r]   = 0.0f;
    if (idx >= BT * H_) emm[r] = -1;
  }

  __syncthreads();

  for (int t = 0; t < T_; ++t) {
    // ---- gather A fragments (h, 16x32 f16 per ISA A layout) ----
    union AU { v16h h; unsigned u[8]; };
    AU A0, A1;
    #pragma unroll
    for (int v = 0; v < 8; ++v) {
      int kl = ((v < 4) ? 0 : 16) + (hi ? 8 : 0) + 2 * (v & 3);
      A0.u[v] = *(const unsigned*)(&h16[m * HS + kl]);        // K 0..31
      A1.u[v] = *(const unsigned*)(&h16[m * HS + 32 + kl]);   // K 32..63
    }

    // ---- recurrent GEMM: 2 K-steps per gate tile ----
    #pragma unroll
    for (int tt = 0; tt < 2; ++tt) {
      int tile = tt * NW + w;           // wave-uniform predicate
      if (tile < 13) {
        v8f acc = {};
        acc = __builtin_amdgcn_wmma_f32_16x16x32_f16(
            false, A0.h, false, Bf[tt][0].h, (short)0, acc, false, false);
        acc = __builtin_amdgcn_wmma_f32_16x16x32_f16(
            false, A1.h, false, Bf[tt][1].h, (short)0, acc, false, false);
        // store D transposed: gates_t[n][m];  VGPR r -> m = r + 8*hi
        float* gp = &gates_t[(tile * 16 + m) * GS + hi * 8];
        v4f lo4 = { acc[0], acc[1], acc[2], acc[3] };
        v4f hi4 = { acc[4], acc[5], acc[6], acc[7] };
        *(v4f*)(gp)     = lo4;
        *(v4f*)(gp + 4) = hi4;
      }
    }

    __syncthreads();

    // ---- LSTM cell update (fp32), h re-emitted as f16 for next A ----
    #pragma unroll
    for (int r = 0; r < 4; ++r) {
      if (emm[r] >= 0) {
        int mm = emm[r], jj = ejj[r];
        float xv = x_sh[mm * XS + t];
        float gi = gates_t[(jj          ) * GS + mm] + xv * Wih_s[jj      ] + bias_s[jj      ];
        float gf = gates_t[(H_   + jj   ) * GS + mm] + xv * Wih_s[H_ + jj ] + bias_s[H_ + jj ];
        float gg = gates_t[(2*H_ + jj   ) * GS + mm] + xv * Wih_s[2*H_+jj ] + bias_s[2*H_+jj ];
        float go = gates_t[(3*H_ + jj   ) * GS + mm] + xv * Wih_s[3*H_+jj ] + bias_s[3*H_+jj ];
        float iv = sigmoidf_(gi);
        float fv = sigmoidf_(gf);
        float gv = tanhf_(gg);
        float ov = sigmoidf_(go);
        float cn = fv * c[r] + iv * gv;
        c[r] = cn;
        float hn = ov * tanhf_(cn);
        h16[mm * HS + jj] = (_Float16)hn;
        if (t == T_ - 1) h32[mm * H32S + jj] = hn;
      }
    }

    __syncthreads();
  }

  // ---- FC head on last hidden state: out[b][o] = h.fcw[o] + fcb[o] ----
  if (tid < BT * 3) {
    int mm = tid / 3, o = tid % 3;
    float s = fcb[o];
    #pragma unroll 10
    for (int j = 0; j < H_; ++j) s += h32[mm * H32S + j] * fcw[o * H_ + j];
    out[(b0 + mm) * 3 + o] = s;
  }
}

extern "C" void kernel_launch(void* const* d_in, const int* in_sizes, int n_in,
                              void* d_out, int out_size, void* d_ws, size_t ws_size,
                              hipStream_t stream) {
  (void)in_sizes; (void)n_in; (void)d_ws; (void)ws_size; (void)out_size;
  const float* x   = (const float*)d_in[0];
  const float* Wih = (const float*)d_in[1];
  const float* Whh = (const float*)d_in[2];
  const float* bih = (const float*)d_in[3];
  const float* bhh = (const float*)d_in[4];
  const float* fcw = (const float*)d_in[5];
  const float* fcb = (const float*)d_in[6];
  float* out = (float*)d_out;

  dim3 grid(B_ / BT);   // 128 blocks
  dim3 block(NT);       // 256 threads = 8 waves
  hipLaunchKernelGGL(lstm_wmma_kernel, grid, block, 0, stream,
                     x, Wih, Whh, bih, bhh, fcw, fcb, out);
}
